// PointMambaBlock_395136991479
// MI455X (gfx1250) — matrix operand order repaired and checked by reading, not compiled, not run on hardware
//
#include <hip/hip_runtime.h>
#include <hip/hip_bf16.h>
#include <math.h>

typedef _Float16 v16h __attribute__((ext_vector_type(16)));
typedef _Float16 v8h  __attribute__((ext_vector_type(8)));
typedef float    v8f  __attribute__((ext_vector_type(8)));

#define BB 4
#define LL 2048
#define CCH 512
#define NN (BB*LL)          // 8192
#define DIN_ 1024
#define DS_ 16
#define DCONV_ 4
#define DTR_ 32
#define EPSF 1e-5f
#define NCH 16              // scan chunks per batch
#define CLEN 128            // chunk length (NCH*CLEN == LL)

// ---------------- CDNA5 async global->LDS copy (ASYNCcnt path) ----------------
#if defined(__has_builtin)
#if __has_builtin(__builtin_amdgcn_global_load_async_to_lds_b128) && \
    __has_builtin(__builtin_amdgcn_s_wait_asynccnt)
#define HAVE_ASYNC_LDS 1
#endif
#endif

#ifdef HAVE_ASYNC_LDS
// builtin expects pointers to int4 (GCC vector_size(16)) in AS1 (global) / AS3 (LDS)
typedef int v4i_gcc __attribute__((vector_size(16)));
typedef __attribute__((address_space(1))) v4i_gcc gv4i;
typedef __attribute__((address_space(3))) v4i_gcc lv4i;

__device__ __forceinline__ void async_cp16(const _Float16* gsrc, _Float16* ldst) {
  __builtin_amdgcn_global_load_async_to_lds_b128(
      (gv4i*)(unsigned long long)(const void*)gsrc,
      (lv4i*)(unsigned int)(unsigned long long)(void*)ldst,
      0, 0);
}
#define WAIT_ASYNC() __builtin_amdgcn_s_wait_asynccnt(0)
#else
#define WAIT_ASYNC()
#endif

// ---------------- generic f32 -> f16 convert ----------------
__global__ void cvt_f32_to_f16(const float* __restrict__ src, _Float16* __restrict__ dst, int n) {
  int i = blockIdx.x * blockDim.x + threadIdx.x;
  int stride = gridDim.x * blockDim.x;
  for (; i < n; i += stride) dst[i] = (_Float16)src[i];
}

// ---------------- WMMA GEMM:  out = epilogue( A[rowmap] @ W^T ) ----------------
// A: (M,K) f16 row-major (rows optionally gathered through rowmap)
// W: (Nout,K) f16 row-major  ->  C[m,n] = sum_k A[m,k]*W[n,k]
// Block tile 128x64, K-step 32, 8 wave32s (4x2), each wave 32x32 (2x2 WMMA frags).
// Double-buffered LDS; async global->LDS copies when available.
// MODE 0: f32 store | 1: BN+ReLU -> f16 | 2: f16 store | 3: softplus(+bias) -> f32
// MODE 4: BN + identity + ReLU -> f32
template<int MODE>
__global__ __launch_bounds__(256) void gemm_wmma(
    const _Float16* __restrict__ A, const _Float16* __restrict__ W,
    const int* __restrict__ rowmap, int M, int Nout, int K,
    float* __restrict__ outf, _Float16* __restrict__ outh,
    const float* __restrict__ gam, const float* __restrict__ bet,
    const float* __restrict__ mu,  const float* __restrict__ var,
    const float* __restrict__ bias, const float* __restrict__ ident)
{
  __shared__ __align__(16) _Float16 sA[2][128 * 40];
  __shared__ __align__(16) _Float16 sB[2][64 * 40];

  const int tid  = threadIdx.x;
  const int lane = tid & 31;
  const int wave = tid >> 5;
  const int wm = wave >> 1;          // 0..3  (M direction)
  const int wn = wave & 1;           // 0..1  (N direction)
  const int hi = lane >> 4;          // half-wave selector
  const int lq = lane & 15;
  const int rowBase = blockIdx.y * 128;
  const int colBase = blockIdx.x * 64;

  // hoist gathered A row indices and W row index for the tile loaders
  int ar[2];
  #pragma unroll
  for (int i = 0; i < 2; ++i) {
    int chunk = tid + 256 * i;       // 0..511
    int r = chunk >> 2;              // 0..127
    int gr = rowBase + r;
    ar[i] = rowmap ? rowmap[gr] : gr;
  }
  const int wrow = colBase + (tid >> 2);

  // tile stager: 128x32 A halves + 64x32 W halves, 16B chunks
  auto issue_tile = [&](int k0, int buf) {
    #pragma unroll
    for (int i = 0; i < 2; ++i) {
      int chunk = tid + 256 * i;
      int r  = chunk >> 2;
      int kc = (chunk & 3) * 8;
      const _Float16* g = A + (size_t)ar[i] * K + k0 + kc;
      _Float16* l = &sA[buf][r * 40 + kc];
#ifdef HAVE_ASYNC_LDS
      async_cp16(g, l);
#else
      *(v8h*)l = *(const v8h*)g;
#endif
    }
    {
      int kc = (tid & 3) * 8;
      const _Float16* g = W + (size_t)wrow * K + k0 + kc;
      _Float16* l = &sB[buf][(tid >> 2) * 40 + kc];
#ifdef HAVE_ASYNC_LDS
      async_cp16(g, l);
#else
      *(v8h*)l = *(const v8h*)g;
#endif
    }
#ifndef HAVE_ASYNC_LDS
    if (k0 + 32 < K) {  // prefetch next K tile (global_prefetch_b8)
      __builtin_prefetch(A + (size_t)ar[0] * K + k0 + 32, 0, 3);
      __builtin_prefetch(W + (size_t)wrow * K + k0 + 32, 0, 3);
    }
#endif
  };

  v8f zero = 0.f;
  v8f acc[2][2];
  acc[0][0] = zero; acc[0][1] = zero; acc[1][0] = zero; acc[1][1] = zero;

  issue_tile(0, 0);
  int cur = 0;
  for (int k0 = 0; k0 < K; k0 += 32) {
    WAIT_ASYNC();        // this wave's pending tile landed in LDS
    __syncthreads();     // all waves' tiles visible; prev buffer fully consumed
    if (k0 + 32 < K) issue_tile(k0 + 32, cur ^ 1);

    // ---- load fragments per ISA VGPR layouts ----
    v16h af[2], bf[2];
    #pragma unroll
    for (int mi = 0; mi < 2; ++mi) {
      int m = wm * 32 + mi * 16 + lq;                      // A: row m = lane&15
      v8h lo = *(const v8h*)&sA[cur][m * 40 + 8 * hi];     // K = 8*hi + (0..7)
      v8h hh = *(const v8h*)&sA[cur][m * 40 + 16 + 8 * hi];// K = 16 + 8*hi + (0..7)
      #pragma unroll
      for (int e = 0; e < 8; ++e) { af[mi][e] = lo[e]; af[mi][e + 8] = hh[e]; }
    }
    #pragma unroll
    for (int ni = 0; ni < 2; ++ni) {
      int n = wn * 32 + ni * 16 + lq;                      // B: col n = lane&15
      v8h lo = *(const v8h*)&sB[cur][n * 40 + 8 * hi];     // K = 8*hi + (0..7)
      v8h hh = *(const v8h*)&sB[cur][n * 40 + 16 + 8 * hi];// K = 16 + 8*hi + (0..7)
      #pragma unroll
      for (int e = 0; e < 8; ++e) { bf[ni][e] = lo[e]; bf[ni][e + 8] = hh[e]; }
    }
    #pragma unroll
    for (int mi = 0; mi < 2; ++mi)
      #pragma unroll
      for (int ni = 0; ni < 2; ++ni)
        acc[mi][ni] = __builtin_amdgcn_wmma_f32_16x16x32_f16(
            false, af[mi], false, bf[ni], (short)0, acc[mi][ni], false, false);
    cur ^= 1;
  }

  // ---- epilogue ----
  #pragma unroll
  for (int ni = 0; ni < 2; ++ni) {
    const int col = colBase + wn * 32 + ni * 16 + lq;
    float scale = 1.f, shift = 0.f, bcol = 0.f;
    if (MODE == 1 || MODE == 4) {
      float s = gam[col] * rsqrtf(var[col] + EPSF);
      scale = s; shift = bet[col] - mu[col] * s;
    }
    if (MODE == 3) bcol = bias[col];
    #pragma unroll
    for (int mi = 0; mi < 2; ++mi) {
      #pragma unroll
      for (int r = 0; r < 8; ++r) {
        int row = rowBase + wm * 32 + mi * 16 + r + 8 * hi;
        float v = acc[mi][ni][r];
        size_t o = (size_t)row * Nout + col;
        if (MODE == 0) {
          outf[o] = v;
        } else if (MODE == 1) {
          v = v * scale + shift; v = v > 0.f ? v : 0.f;
          outh[o] = (_Float16)v;
        } else if (MODE == 2) {
          outh[o] = (_Float16)v;
        } else if (MODE == 3) {
          v += bcol;
          outf[o] = (v > 20.f) ? v : log1pf(__expf(v));
        } else {  // MODE 4
          v = v * scale + shift + ident[o];
          outf[o] = v > 0.f ? v : 0.f;
        }
      }
    }
  }
}

// ---------------- depthwise causal conv (k=4) + SiLU ----------------
__global__ void conv_silu(const float* __restrict__ xz, const float* __restrict__ cw,
                          const float* __restrict__ cb, float* __restrict__ xc,
                          _Float16* __restrict__ xch) {
  int idx = blockIdx.x * blockDim.x + threadIdx.x;   // over NN*DIN
  if (idx >= NN * DIN_) return;
  int d   = idx & (DIN_ - 1);
  int row = idx >> 10;                                // b*LL + t
  int t   = row & (LL - 1);
  float accv = cb[d];
  #pragma unroll
  for (int j = 0; j < DCONV_; ++j) {
    int back = DCONV_ - 1 - j;
    if (t >= back)
      accv += cw[d * DCONV_ + j] * xz[(size_t)(row - back) * (2 * DIN_) + d];
  }
  float s = accv / (1.f + __expf(-accv));
  xc[idx]  = s;
  xch[idx] = (_Float16)s;
}

// ---------------- slice dbl[:, :32] -> f16 dt_low ----------------
__global__ void slice_dtlow(const float* __restrict__ dbl, _Float16* __restrict__ dlo) {
  int idx = blockIdx.x * blockDim.x + threadIdx.x;   // over NN*DTR
  if (idx >= NN * DTR_) return;
  int m = idx >> 5, c = idx & 31;
  dlo[idx] = (_Float16)dbl[(size_t)m * 64 + c];
}

// ======== chunked selective scan: sequential depth 2048 -> 128+16 ========
// h_t[s] = a_t[s]*h_{t-1}[s] + dt_t*u_t*B_t[s],  a_t[s] = exp(dt_t*A[d,s])
// pass1: per chunk, from h=0: P[s]=prod a, F[s]=local final h
__global__ __launch_bounds__(256) void scan_pass1(
    const float* __restrict__ dt, const float* __restrict__ dbl,
    const float* __restrict__ xc, const float* __restrict__ A_log,
    float* __restrict__ P, float* __restrict__ F)
{
  const int bx = blockIdx.x;          // BB*NCH*(DIN/256)
  const int dg = bx & 3;
  const int c  = (bx >> 2) & (NCH - 1);
  const int b  = bx >> 6;
  const int d  = dg * 256 + threadIdx.x;

  float Arow[DS_];
  #pragma unroll
  for (int s = 0; s < DS_; ++s) Arow[s] = -__expf(A_log[d * DS_ + s]);
  float h[DS_], p[DS_];
  #pragma unroll
  for (int s = 0; s < DS_; ++s) { h[s] = 0.f; p[s] = 1.f; }

  __shared__ float BC[DS_];
  for (int t0 = 0; t0 < CLEN; ++t0) {
    const int row = b * LL + c * CLEN + t0;
    __syncthreads();
    if (threadIdx.x < DS_) BC[threadIdx.x] = dbl[(size_t)row * 64 + DTR_ + threadIdx.x];
    __syncthreads();
    float dtv = dt[(size_t)row * DIN_ + d];
    float du  = dtv * xc[(size_t)row * DIN_ + d];
    #pragma unroll
    for (int s = 0; s < DS_; ++s) {
      float a = __expf(dtv * Arow[s]);
      h[s] = a * h[s] + du * BC[s];
      p[s] *= a;
    }
  }
  size_t o = ((size_t)(b * NCH + c) * DIN_ + d) * DS_;
  #pragma unroll
  for (int s = 0; s < DS_; ++s) { P[o + s] = p[s]; F[o + s] = h[s]; }
}

// pass2: tiny sequential combine over chunks -> initial state H0 per chunk
__global__ void scan_pass2(const float* __restrict__ P, const float* __restrict__ F,
                           float* __restrict__ H0) {
  int idx = blockIdx.x * blockDim.x + threadIdx.x;  // over BB*DIN*DS = 65536
  if (idx >= BB * DIN_ * DS_) return;
  int s = idx & (DS_ - 1);
  int d = (idx >> 4) & (DIN_ - 1);
  int b = idx >> 14;
  float h = 0.f;
  for (int c = 0; c < NCH; ++c) {
    size_t o = ((size_t)(b * NCH + c) * DIN_ + d) * DS_ + s;
    H0[o] = h;
    h = P[o] * h + F[o];
  }
}

// pass3: rescan from H0, fuse y = (sum h*C + u*D) * silu(z) -> f16
__global__ __launch_bounds__(256) void scan_pass3(
    const float* __restrict__ dt, const float* __restrict__ dbl,
    const float* __restrict__ xc, const float* __restrict__ A_log,
    const float* __restrict__ H0, const float* __restrict__ xz,
    const float* __restrict__ Dp, _Float16* __restrict__ yh)
{
  const int bx = blockIdx.x;
  const int dg = bx & 3;
  const int c  = (bx >> 2) & (NCH - 1);
  const int b  = bx >> 6;
  const int d  = dg * 256 + threadIdx.x;

  float Arow[DS_];
  #pragma unroll
  for (int s = 0; s < DS_; ++s) Arow[s] = -__expf(A_log[d * DS_ + s]);
  float h[DS_];
  size_t oh = ((size_t)(b * NCH + c) * DIN_ + d) * DS_;
  #pragma unroll
  for (int s = 0; s < DS_; ++s) h[s] = H0[oh + s];
  const float dd = Dp[d];

  __shared__ float BC[2 * DS_];
  for (int t0 = 0; t0 < CLEN; ++t0) {
    const int row = b * LL + c * CLEN + t0;
    __syncthreads();
    if (threadIdx.x < 2 * DS_)
      BC[threadIdx.x] = dbl[(size_t)row * 64 + DTR_ + threadIdx.x];
    __syncthreads();
    float dtv = dt[(size_t)row * DIN_ + d];
    float u   = xc[(size_t)row * DIN_ + d];
    float du  = dtv * u;
    float y = 0.f;
    #pragma unroll
    for (int s = 0; s < DS_; ++s) {
      float a = __expf(dtv * Arow[s]);
      h[s] = a * h[s] + du * BC[s];
      y += h[s] * BC[DS_ + s];
    }
    float z = xz[(size_t)row * (2 * DIN_) + DIN_ + d];
    float sil = z / (1.f + __expf(-z));
    yh[(size_t)row * DIN_ + d] = (_Float16)((y + u * dd) * sil);
  }
}

// ---------------- launch ----------------
extern "C" void kernel_launch(void* const* d_in, const int* in_sizes, int n_in,
                              void* d_out, int out_size, void* d_ws, size_t ws_size,
                              hipStream_t stream) {
  (void)in_sizes; (void)n_in; (void)out_size; (void)ws_size;
  const float* x        = (const float*)d_in[1];
  const int*   order    = (const int*)d_in[3];
  const int*   inv      = (const int*)d_in[4];
  const float* W_in     = (const float*)d_in[5];
  const float* g_in     = (const float*)d_in[6];
  const float* b_in     = (const float*)d_in[7];
  const float* m_in     = (const float*)d_in[8];
  const float* v_in     = (const float*)d_in[9];
  const float* in_proj  = (const float*)d_in[10];
  const float* conv_w   = (const float*)d_in[11];
  const float* conv_b   = (const float*)d_in[12];
  const float* x_proj   = (const float*)d_in[13];
  const float* dt_proj  = (const float*)d_in[14];
  const float* dt_pb    = (const float*)d_in[15];
  const float* A_log    = (const float*)d_in[16];
  const float* Dp       = (const float*)d_in[17];
  const float* out_proj = (const float*)d_in[18];
  const float* W_out    = (const float*)d_in[19];
  const float* g_o      = (const float*)d_in[20];
  const float* b_o      = (const float*)d_in[21];
  const float* m_o      = (const float*)d_in[22];
  const float* v_o      = (const float*)d_in[23];
  float* out = (float*)d_out;

  char* wsp = (char*)d_ws;
  auto carve = [&](size_t bytes) -> char* {
    char* p = wsp;
    wsp += (bytes + 255) & ~(size_t)255;
    return p;
  };
  _Float16* x16   = (_Float16*)carve((size_t)NN * CCH * 2);
  _Float16* Wi16  = (_Float16*)carve((size_t)CCH * CCH * 2);
  _Float16* Pi16  = (_Float16*)carve((size_t)2 * DIN_ * CCH * 2);
  _Float16* Xp16  = (_Float16*)carve((size_t)(DTR_ + 2 * DS_) * DIN_ * 2);
  _Float16* Dt16  = (_Float16*)carve((size_t)DIN_ * DTR_ * 2);
  _Float16* Op16  = (_Float16*)carve((size_t)CCH * DIN_ * 2);
  _Float16* Wo16  = (_Float16*)carve((size_t)CCH * CCH * 2);
  _Float16* h16   = (_Float16*)carve((size_t)NN * CCH * 2);
  float*    xz    = (float*)   carve((size_t)NN * 2 * DIN_ * 4);
  float*    xc    = (float*)   carve((size_t)NN * DIN_ * 4);
  _Float16* xc16  = (_Float16*)carve((size_t)NN * DIN_ * 2);
  float*    dbl   = (float*)   carve((size_t)NN * 64 * 4);
  _Float16* dlo16 = (_Float16*)carve((size_t)NN * DTR_ * 2);
  float*    dt    = (float*)   carve((size_t)NN * DIN_ * 4);
  float*    Pc    = (float*)   carve((size_t)BB * NCH * DIN_ * DS_ * 4);
  float*    Fc    = (float*)   carve((size_t)BB * NCH * DIN_ * DS_ * 4);
  float*    H0    = (float*)   carve((size_t)BB * NCH * DIN_ * DS_ * 4);
  _Float16* y16   = (_Float16*)carve((size_t)NN * DIN_ * 2);
  _Float16* yo16  = (_Float16*)carve((size_t)NN * CCH * 2);

  // --- convert inputs/weights to f16 ---
  auto cvt = [&](const float* s, _Float16* d, int n) {
    int blocks = (n + 255) / 256; if (blocks > 2048) blocks = 2048;
    cvt_f32_to_f16<<<blocks, 256, 0, stream>>>(s, d, n);
  };
  cvt(x, x16, NN * CCH);
  cvt(W_in, Wi16, CCH * CCH);
  cvt(in_proj, Pi16, 2 * DIN_ * CCH);
  cvt(x_proj, Xp16, (DTR_ + 2 * DS_) * DIN_);
  cvt(dt_proj, Dt16, DIN_ * DTR_);
  cvt(out_proj, Op16, CCH * DIN_);
  cvt(W_out, Wo16, CCH * CCH);

  // --- GEMM1: h = f16(relu(bn_in(x @ W_in^T))) ---
  gemm_wmma<1><<<dim3(CCH / 64, NN / 128), 256, 0, stream>>>(
      x16, Wi16, nullptr, NN, CCH, CCH, nullptr, h16,
      g_in, b_in, m_in, v_in, nullptr, nullptr);

  // --- GEMM2: xz = h[order] @ in_proj^T  (f32, 8192x2048) ---
  gemm_wmma<0><<<dim3(2 * DIN_ / 64, NN / 128), 256, 0, stream>>>(
      h16, Pi16, order, NN, 2 * DIN_, CCH, xz, nullptr,
      nullptr, nullptr, nullptr, nullptr, nullptr, nullptr);

  // --- depthwise causal conv + SiLU ---
  conv_silu<<<(NN * DIN_) / 256, 256, 0, stream>>>(xz, conv_w, conv_b, xc, xc16);

  // --- GEMM3: dbl = xc @ x_proj^T  (f32, 8192x64) ---
  gemm_wmma<0><<<dim3(1, NN / 128), 256, 0, stream>>>(
      xc16, Xp16, nullptr, NN, 64, DIN_, dbl, nullptr,
      nullptr, nullptr, nullptr, nullptr, nullptr, nullptr);

  // --- dt_low slice -> f16 ---
  slice_dtlow<<<(NN * DTR_) / 256, 256, 0, stream>>>(dbl, dlo16);

  // --- GEMM4: dt = softplus(dt_low @ dt_proj^T + b)  (f32, 8192x1024) ---
  gemm_wmma<3><<<dim3(DIN_ / 64, NN / 128), 256, 0, stream>>>(
      dlo16, Dt16, nullptr, NN, DIN_, DTR_, dt, nullptr,
      nullptr, nullptr, nullptr, nullptr, dt_pb, nullptr);

  // --- chunked selective scan (3 passes) + fused gate ---
  scan_pass1<<<BB * NCH * (DIN_ / 256), 256, 0, stream>>>(dt, dbl, xc, A_log, Pc, Fc);
  scan_pass2<<<(BB * DIN_ * DS_) / 256, 256, 0, stream>>>(Pc, Fc, H0);
  scan_pass3<<<BB * NCH * (DIN_ / 256), 256, 0, stream>>>(dt, dbl, xc, A_log,
                                                          H0, xz, Dp, y16);

  // --- GEMM5: yo = y @ out_proj^T -> f16 ---
  gemm_wmma<2><<<dim3(CCH / 64, NN / 128), 256, 0, stream>>>(
      y16, Op16, nullptr, NN, CCH, DIN_, nullptr, yo16,
      nullptr, nullptr, nullptr, nullptr, nullptr, nullptr);

  // --- GEMM6: out = relu(bn_out(yo[inv] @ W_out^T) + x) ---
  gemm_wmma<4><<<dim3(CCH / 64, NN / 128), 256, 0, stream>>>(
      yo16, Wo16, inv, NN, CCH, CCH, out, nullptr,
      g_o, b_o, m_o, v_o, nullptr, x);
}
